// PatchLoss_22746146800083
// MI455X (gfx1250) — matrix operand order repaired and checked by reading, not compiled, not run on hardware
//
#include <hip/hip_runtime.h>

// MI455X (gfx1250) fused contrastive-loss kernel.
//
// Roofline: 2*8192^2*256 = 34.4 GFLOP matmul vs 16.8MB of input (0.7us @
// 23.3TB/s HBM) -> compute bound on the matrix pipe. Strategy:
//  1) Normalize rows in fp32 (exact), pack to BF16 WMMA fragment layout.
//  2) Fused 8192x8192x256 BF16 WMMA GEMM + online sum-of-exp (flash-style),
//     never materializing the 256MB logit matrix. Fixed shift C = 1/0.07
//     is a provable upper bound on |cos|/T, so no running max is needed and
//     the -10 diagonal underflows to exactly 0 (matches fp32 reference).
//  3) Tiny combine kernel: loss = C + log(sum) - l_pos/T.
//
// Round-3 changes: (a) hard scheduling barrier between the B-fragment LDS
// loads and the WMMAs so the compiler cannot sink the loads back into the
// WMMA chain (previously every WMMA stalled on s_wait_dscnt 0); (b) diagonal
// masking hoisted behind a wave-uniform branch (only 1 of 64 tiles can hold
// the diagonal); (c) prefetch locality hint raised.
//
// Workspace layout (N=8192): Qfrag 4MB | Kfrag 4MB | lpos 32KB | partial 256KB.

#define D256 256
#define TEMP_INV 14.285714285714286f /* 1/0.07 ; also the logsumexp shift C */

typedef __attribute__((ext_vector_type(16))) __bf16 v16bf;
typedef __attribute__((ext_vector_type(8)))  float  v8f;

struct Pack32 { uint4 lo, hi; };   // 32 bytes == one v16bf fragment per lane

__device__ __forceinline__ void sched_fence() {
#if __has_builtin(__builtin_amdgcn_sched_barrier)
  __builtin_amdgcn_sched_barrier(0);       // nothing crosses this point
#else
  asm volatile("" ::: "memory");
#endif
}

__device__ __forceinline__ unsigned f2bf1(float f) {
  unsigned u = __builtin_bit_cast(unsigned, f);
  return (u + 0x7FFFu + ((u >> 16) & 1u)) >> 16;       // round-to-nearest-even
}
__device__ __forceinline__ unsigned pk2(float a, float b) {
  return f2bf1(a) | (f2bf1(b) << 16);
}

// ---------------------------------------------------------------------------
// Kernel 1: per-row L2 normalize (fp32), emit BF16 WMMA fragments + l_pos/T.
// One wave per row: lane L owns the 8 floats at column 8*L.
// A-fragment (Q, 16x32 bf16): lane m      elems0-7 = K[0..7],  elems8-15 = K[16..23]
//                             lane m+16   elems0-7 = K[8..15], elems8-15 = K[24..31]
// B-fragment (K, 32x16 bf16): lane n      elems0-7 = K[0..7],  elems8-15 = K[16..23]
//                             lane n+16   elems0-7 = K[8..15], elems8-15 = K[24..31]
// ---------------------------------------------------------------------------
__global__ __launch_bounds__(256)
void normalize_pack_kernel(const float* __restrict__ q,
                           const float* __restrict__ k,
                           uint4* __restrict__ qfrag,
                           uint4* __restrict__ kfrag,
                           float* __restrict__ lpos,
                           int N)
{
  const int lane = threadIdx.x & 31;
  const int wave = threadIdx.x >> 5;
  const int row  = blockIdx.x * 8 + wave;
  if (row >= N) return;                       // whole-wave uniform guard

  const float4* q4 = (const float4*)(q + (size_t)row * D256) + lane * 2;
  const float4* k4 = (const float4*)(k + (size_t)row * D256) + lane * 2;
  float4 qa = q4[0], qb = q4[1];
  float4 ka = k4[0], kb = k4[1];

  float qq = qa.x*qa.x + qa.y*qa.y + qa.z*qa.z + qa.w*qa.w
           + qb.x*qb.x + qb.y*qb.y + qb.z*qb.z + qb.w*qb.w;
  float kk = ka.x*ka.x + ka.y*ka.y + ka.z*ka.z + ka.w*ka.w
           + kb.x*kb.x + kb.y*kb.y + kb.z*kb.z + kb.w*kb.w;
  float qk = qa.x*ka.x + qa.y*ka.y + qa.z*ka.z + qa.w*ka.w
           + qb.x*kb.x + qb.y*kb.y + qb.z*kb.z + qb.w*kb.w;

  #pragma unroll
  for (int off = 16; off; off >>= 1) {        // wave32 butterfly reduce
    qq += __shfl_xor(qq, off, 32);
    kk += __shfl_xor(kk, off, 32);
    qk += __shfl_xor(qk, off, 32);
  }
  const float rq = rsqrtf(qq), rk = rsqrtf(kk);
  if (lane == 0) lpos[row] = qk * rq * rk * TEMP_INV;

  uint4 qo, ko;
  qo.x = pk2(qa.x*rq, qa.y*rq);  qo.y = pk2(qa.z*rq, qa.w*rq);
  qo.z = pk2(qb.x*rq, qb.y*rq);  qo.w = pk2(qb.z*rq, qb.w*rq);
  ko.x = pk2(ka.x*rk, ka.y*rk);  ko.y = pk2(ka.z*rk, ka.w*rk);
  ko.z = pk2(kb.x*rk, kb.y*rk);  ko.w = pk2(kb.z*rk, kb.w*rk);

  const int t = row >> 4, m = row & 15;
  const int b = lane >> 2;                    // which 32-wide K block (0..7)
  const int c = lane & 3;                     // 8-wide sub-chunk within block
  // A-layout: chunk c -> {destLane = m + (c&1)*16, elemOff = (c>>1)*8}
  const int qidx = ((t * 8 + b) * 32 + (m + ((c & 1) << 4))) * 2 + (c >> 1);
  qfrag[qidx] = qo;
  // B-layout: chunk c -> {destLane = m + (c&1)*16, elemOff = (c>>1)*8}
  const int kidx = ((t * 8 + b) * 32 + (m + ((c & 1) << 4))) * 2 + (c >> 1);
  kfrag[kidx] = ko;
}

// ---------------------------------------------------------------------------
// Kernel 2: fused BF16 WMMA GEMM + online sum-of-exp.
// Grid: (N/16/8 row-groups) x (8 column chunks). Block = 256 threads = 8 waves.
// Each wave owns one 16-row Q tile (8 fragments pinned in 64 VGPRs) and sweeps
// tilesPerChunk K tiles; the 8 waves share each 8KB K tile via double-buffered
// LDS. 8 v_wmma_f32_16x16x32_bf16 per output tile (K = 256).
// ---------------------------------------------------------------------------
__global__ __launch_bounds__(256)
void wmma_lse_kernel(const uint4* __restrict__ qfrag,
                     const uint4* __restrict__ kfrag,
                     float* __restrict__ partial,
                     int tilesPerChunk)
{
  __shared__ __attribute__((aligned(16))) uint4 lds[2][512];   // 2 x 8KB K tiles

  const int lane    = threadIdx.x & 31;
  const int wave    = threadIdx.x >> 5;
  const int rowTile = blockIdx.x * 8 + wave;
  const int chunk   = blockIdx.y;
  const int rowBase = rowTile * 16;
  const int ct0     = chunk * tilesPerChunk;

  // Pin this wave's Q tile fragments in registers (8 x v16bf = 64 VGPRs).
  v16bf qf[8];
  #pragma unroll
  for (int kb = 0; kb < 8; ++kb) {
    const int idx = ((rowTile * 8 + kb) * 32 + lane) * 2;
    Pack32 p; p.lo = qfrag[idx]; p.hi = qfrag[idx + 1];
    qf[kb] = __builtin_bit_cast(v16bf, p);
  }

  // Preload first K tile into buffer 0 (256 threads x 2 uint4 = 8KB).
  {
    const uint4* src = kfrag + (size_t)ct0 * 512;
    lds[0][threadIdx.x]       = src[threadIdx.x];
    lds[0][threadIdx.x + 256] = src[threadIdx.x + 256];
  }
  __syncthreads();

  float sums[8] = {0.f, 0.f, 0.f, 0.f, 0.f, 0.f, 0.f, 0.f};
  const int m0 = (lane >> 4) << 3;            // C/D layout: lanes16-31 -> M+8
  const int n  = lane & 15;

  for (int i = 0; i < tilesPerChunk; ++i) {
    const int cur = i & 1;

    // Phase 1: issue next tile's GLOBAL loads into registers (in flight for
    // the entire tile compute; ds_stores deferred to phase 4).
    uint4 stage0, stage1;
    const bool haveNext = (i + 1 < tilesPerChunk);
    if (haveNext) {
      const uint4* src = kfrag + (size_t)(ct0 + i + 1) * 512;
      stage0 = src[threadIdx.x];
      stage1 = src[threadIdx.x + 256];
      if (i + 2 < tilesPerChunk)              // warm caches two tiles ahead
        __builtin_prefetch(kfrag + (size_t)(ct0 + i + 2) * 512 + threadIdx.x * 2, 0, 3);
    }

    // Phase 2: preload ALL 8 B fragments into distinct VGPRs. The sched
    // fence pins the 16 ds_load_b128s before the first WMMA so they all
    // stay live (distinct registers) and the dscnt waits pipeline, instead
    // of each WMMA stalling on s_wait_dscnt 0.
    v16bf kf[8];
    #pragma unroll
    for (int kb = 0; kb < 8; ++kb) {
      Pack32 p;
      p.lo = lds[cur][(kb * 32 + lane) * 2];
      p.hi = lds[cur][(kb * 32 + lane) * 2 + 1];
      kf[kb] = __builtin_bit_cast(v16bf, p);
    }
    sched_fence();

    // Phase 3: 8 chained WMMAs (K = 256).
    v8f acc = {0.f, 0.f, 0.f, 0.f, 0.f, 0.f, 0.f, 0.f};
    #pragma unroll
    for (int kb = 0; kb < 8; ++kb) {
      acc = __builtin_amdgcn_wmma_f32_16x16x32_bf16(
          /*neg_a=*/false, qf[kb], /*neg_b=*/false, kf[kb],
          /*c_mod=*/(short)0, acc, /*reuse_a=*/false, /*reuse_b=*/false);
    }

    // Online sum of exp with fixed shift C = 1/T (provable bound on cos/T).
    if (ct0 + i != rowTile) {                  // wave-uniform: no diagonal here
      #pragma unroll
      for (int v = 0; v < 8; ++v)
        sums[v] += __expf(acc[v] * TEMP_INV - TEMP_INV);
    } else {                                   // 1 of 64 tiles: mask diagonal
      const int gcol = (ct0 + i) * 16 + n;
      #pragma unroll
      for (int v = 0; v < 8; ++v) {
        float e = __expf(acc[v] * TEMP_INV - TEMP_INV);
        if (rowBase + m0 + v == gcol) e = 0.0f; // diag -10/T underflows to 0
        sums[v] += e;
      }
    }

    // Phase 4: commit the staged next tile to the other LDS buffer.
    if (haveNext) {
      lds[cur ^ 1][threadIdx.x]       = stage0;
      lds[cur ^ 1][threadIdx.x + 256] = stage1;
    }
    __syncthreads();   // buffer cur is free to be refilled next iteration
  }

  // Reduce each row's 16 lanes (lanes 0-15 and 16-31 hold different M).
  #pragma unroll
  for (int v = 0; v < 8; ++v) {
    float s = sums[v];
    s += __shfl_xor(s, 1, 32);
    s += __shfl_xor(s, 2, 32);
    s += __shfl_xor(s, 4, 32);
    s += __shfl_xor(s, 8, 32);
    sums[v] = s;
  }
  if (n == 0) {                                // lanes 0 and 16 write
    #pragma unroll
    for (int v = 0; v < 8; ++v)
      partial[(size_t)(rowBase + m0 + v) * 8 + chunk] = sums[v];
  }
}

// ---------------------------------------------------------------------------
// Kernel 3: combine 8 chunk partials + positive logit into the final loss.
// loss = logsumexp(out) - out[0] = C + log(sum_exp) - lpos/T
// ---------------------------------------------------------------------------
__global__ __launch_bounds__(256)
void combine_kernel(const float* __restrict__ partial,
                    const float* __restrict__ lpos,
                    float* __restrict__ out, int N)
{
  const int r = blockIdx.x * blockDim.x + threadIdx.x;
  if (r >= N) return;
  float t = 0.f;
  #pragma unroll
  for (int c = 0; c < 8; ++c) t += partial[(size_t)r * 8 + c];
  const float lp = lpos[r];
  t += __expf(lp - TEMP_INV);                  // the l_pos column (index 0)
  out[r] = TEMP_INV + __logf(t) - lp;
}

extern "C" void kernel_launch(void* const* d_in, const int* in_sizes, int n_in,
                              void* d_out, int out_size, void* d_ws, size_t ws_size,
                              hipStream_t stream)
{
  const float* q = (const float*)d_in[0];
  const float* k = (const float*)d_in[1];
  const int N = in_sizes[0] / D256;            // 8192

  unsigned char* ws = (unsigned char*)d_ws;
  uint4* qfrag   = (uint4*)ws;                                   // N*256*2 B
  uint4* kfrag   = (uint4*)(ws + (size_t)N * D256 * 2);          // N*256*2 B
  float* lpos    = (float*)(ws + (size_t)N * D256 * 4);          // N*4 B
  float* partial = (float*)(ws + (size_t)N * D256 * 4 + (size_t)N * 4); // N*8*4 B

  // 1) normalize + fragment pack (one wave per row)
  normalize_pack_kernel<<<(N + 7) / 8, 256, 0, stream>>>(q, k, qfrag, kfrag, lpos, N);

  // 2) fused WMMA GEMM + partial logsumexp: 64 row-groups x 8 column chunks
  const int rowGroups     = (N / 16) / 8;      // 64
  const int tilesPerChunk = (N / 16) / 8;      // 64 tiles of 16 cols per chunk
  wmma_lse_kernel<<<dim3(rowGroups, 8), 256, 0, stream>>>(qfrag, kfrag, partial,
                                                          tilesPerChunk);

  // 3) combine partials into the loss
  combine_kernel<<<(N + 255) / 256, 256, 0, stream>>>(partial, lpos, (float*)d_out, N);
}
